// BlockLocalSelfAttention_64269890617615
// MI455X (gfx1250) — matrix operand order repaired and checked by reading, not compiled, not run on hardware
//
#include <hip/hip_runtime.h>

typedef __attribute__((ext_vector_type(16))) _Float16 v16h;
typedef __attribute__((ext_vector_type(8)))  _Float16 v8h;
typedef __attribute__((ext_vector_type(2)))  _Float16 h2;
typedef __attribute__((ext_vector_type(8)))  float    v8f;

#define NB   2
#define NH   16
#define T    4000
#define D    64
#define BLK  128
#define NW   32        /* windows: (4000+96)/128 */
#define NKEYS 385      /* 1 global + 3*128 window keys */
#define KTILE 32
#define NKT  13        /* ceil(385/32) */
#define SCALE 0.125f   /* 1/sqrt(64) */
#define FMIN (-3.402823466e38f)

/* LDS strides (halfs); chosen so fragment reads are 16/32B aligned */
#define KS_STRIDE 80   /* K tile: [32 keys][64 d], row = 160B */
#define VT_STRIDE 48   /* V^T:    [64 d][32 keys], row = 96B  */
#define P_STRIDE  40   /* probs:  [16 rows][32 keys], row = 80B */

// ---------------------------------------------------------------------------
// Main kernel: one workgroup per (window, head, batch). 8 wave32 per block,
// each wave owns 16 query rows. Flash-attention over 13 key tiles of 32 keys
// using v_wmma_f32_16x16x32_f16 for QK^T, P*V, and the softmax row-sums
// (P x ones trick). Row max: packed-f16 ds_swizzle xor reduction
// (4 swizzles + 4 v_pk_max per row-pair, 16 DS ops per tile).
// ---------------------------------------------------------------------------
__global__ __launch_bounds__(256) void bla_main_kernel(
    const float* __restrict__ q, const float* __restrict__ k,
    const float* __restrict__ v, const float* __restrict__ am,
    float* __restrict__ out)
{
    __shared__ __align__(32) _Float16 Ks[32 * KS_STRIDE];
    __shared__ __align__(32) _Float16 Vt[64 * VT_STRIDE];
    __shared__ __align__(32) _Float16 Pb[8 * 16 * P_STRIDE];
    __shared__ float Mv[KTILE];

    const int w   = blockIdx.x;
    const int hh  = blockIdx.y;
    const int nn  = blockIdx.z;
    const int tid = threadIdx.x;
    const int lane = tid & 31;
    const int wv   = tid >> 5;      // wave id 0..7
    const int g    = lane >> 4;     // lane half 0/1
    const int ln   = lane & 15;

    const size_t headBase = ((size_t)(nn * NH + hh)) * T * D;
    const float* kh  = k + headBase;
    const float* vh  = v + headBase;
    const float* amn = am + (size_t)nn * T;

    // ---- Load this wave's 16x64 Q tile as two A fragments (f32 -> f16) ----
    // A order used here (16-bit, 16x32, contiguous k, same as B): lane L: M = L%16; element e: K = 16*(L/16) + e
    v16h aq0, aq1;
    {
        int qpos = w * BLK + wv * 16 + ln;
        int qr = qpos < T ? qpos : T - 1;     // clamp; invalid rows never stored
        const float* qrow = q + headBase + (size_t)qr * D;
#pragma unroll
        for (int fi = 0; fi < 2; ++fi) {
            v16h a;
#pragma unroll
            for (int c = 0; c < 2; ++c) {
                const float* src = qrow + 32 * fi + 16 * g + 8 * c;
#pragma unroll
                for (int e = 0; e < 8; ++e)
                    a[c * 8 + e] = (_Float16)src[e];
            }
            if (fi == 0) aq0 = a; else aq1 = a;
        }
    }

    // constant all-ones B fragment: P(16x32) x ones(32x16) -> row sums of P
    v16h vones;
#pragma unroll
    for (int e = 0; e < 16; ++e) vones[e] = (_Float16)1.0f;

    // ---- flash state: O (16x64 as 4 C frags), o4 = running row-sums, row max
    v8f o0 = {}, o1 = {}, o2 = {}, o3 = {}, o4 = {};
    float mrow[8];
#pragma unroll
    for (int r = 0; r < 8; ++r) mrow[r] = FMIN;

    _Float16* Pw = Pb + wv * 16 * P_STRIDE;

    for (int j = 0; j < NKT; ++j) {
        __syncthreads();   // previous tile fully consumed before restaging
        // ---- stage K tile (row-major) and V tile (transposed) into LDS ----
#pragma unroll
        for (int p = 0; p < 8; ++p) {
            int idx = p * 256 + tid;        // 0..2047
            int ks  = idx >> 6;             // key slot 0..31
            int dd  = idx & 63;
            int kk  = j * KTILE + ks;       // index into 385-key window (+pad)
            int pos; bool valid;
            if (kk == 0)            { pos = 0; valid = true; }               // global key
            else if (kk < NKEYS)    { pos = w * BLK + kk - 129;              // local window
                                      valid = (pos >= 0) && (pos < T); }
            else                    { pos = 0; valid = false; }              // tile pad
            float kvv = 0.f, vvv = 0.f;
            if (valid) {
                size_t off = (size_t)pos * D + dd;
                kvv = kh[off]; vvv = vh[off];
            }
            Ks[ks * KS_STRIDE + dd] = (_Float16)kvv;
            Vt[dd * VT_STRIDE + ks] = (_Float16)vvv;
        }
        if (tid < KTILE) {
            int kk = j * KTILE + tid;
            float mv;
            if (kk == 0) mv = amn[0];                       // global key mask
            else if (kk < NKEYS) {
                int pos = w * BLK + kk - 129;
                mv = (pos > 0 && pos < T) ? amn[pos] : FMIN; // pos==0 excluded locally
            } else mv = FMIN;
            Mv[tid] = mv;
        }
        __syncthreads();

        // ---- prefetch next tile's K/V rows (global_prefetch_b8) ----
        if (j + 1 < NKT) {
            int kk = (j + 1) * KTILE + (tid >> 3);   // 32 key rows, 8 threads each
            int seg = tid & 7;                        // 8 x 32B spots per 256B row
            int pos = (kk == 0) ? 0 : (w * BLK + kk - 129);
            pos = pos < 0 ? 0 : (pos >= T ? T - 1 : pos);
            const float* pf = kh + (size_t)pos * D + seg * 8;
            __builtin_prefetch(pf, 0, 3);
            __builtin_prefetch(vh + (size_t)pos * D + seg * 8, 0, 3);
        }

        // ---- S = Q * K^T (two 16-key halves, accumulate over d in 2 steps) ----
        float s[2][8];
#pragma unroll
        for (int nt = 0; nt < 2; ++nt) {
            // B order used here (16-bit, 32x16, contiguous k, same as A): lane L: N = L%16; element e: K = e + 16*(L/16)
            const _Float16* bbase = Ks + (nt * 16 + ln) * KS_STRIDE + 16 * g;
            v16h b0 = *(const v16h*)(bbase);
            v16h b1 = *(const v16h*)(bbase + 32);
            v8f c = {};
            c = __builtin_amdgcn_wmma_f32_16x16x32_f16(false, aq0, false, b0, (short)0, c, false, false);
            c = __builtin_amdgcn_wmma_f32_16x16x32_f16(false, aq1, false, b1, (short)0, c, false, false);
            float mv = Mv[nt * 16 + ln];
#pragma unroll
            for (int r = 0; r < 8; ++r) s[nt][r] = c[r] * SCALE + mv;
        }

        // ---- row max: packed f16 pairs; xor exchange via ds_swizzle,
        //      combine via v_pk_max_num_f16 (__builtin_elementwise_max).
        //      swizzle imm = (xor_mask<<10) | 0x1f  (group-of-32 mode) ----
        float mnew[8];
#pragma unroll
        for (int rp = 0; rp < 4; ++rp) {
            union { int i; h2 h; } pk, ot;
            pk.h[0] = (_Float16)fmaxf(s[0][2 * rp],     s[1][2 * rp]);
            pk.h[1] = (_Float16)fmaxf(s[0][2 * rp + 1], s[1][2 * rp + 1]);
            ot.i = __builtin_amdgcn_ds_swizzle(pk.i, 0x041f);   // lane ^ 1
            pk.h = __builtin_elementwise_max(pk.h, ot.h);
            ot.i = __builtin_amdgcn_ds_swizzle(pk.i, 0x081f);   // lane ^ 2
            pk.h = __builtin_elementwise_max(pk.h, ot.h);
            ot.i = __builtin_amdgcn_ds_swizzle(pk.i, 0x101f);   // lane ^ 4
            pk.h = __builtin_elementwise_max(pk.h, ot.h);
            ot.i = __builtin_amdgcn_ds_swizzle(pk.i, 0x201f);   // lane ^ 8
            pk.h = __builtin_elementwise_max(pk.h, ot.h);
            mnew[2 * rp]     = fmaxf(mrow[2 * rp],     (float)pk.h[0]);
            mnew[2 * rp + 1] = fmaxf(mrow[2 * rp + 1], (float)pk.h[1]);
        }

        // ---- online rescale + exp; stash P (f16) to LDS in A-friendly layout
#pragma unroll
        for (int r = 0; r < 8; ++r) {
            float mn    = mnew[r];
            float alpha = __expf(mrow[r] - mn);
            mrow[r] = mn;
            float p0 = __expf(s[0][r] - mn);
            float p1 = __expf(s[1][r] - mn);
            o0[r] *= alpha; o1[r] *= alpha; o2[r] *= alpha; o3[r] *= alpha;
            o4[r] *= alpha;
            Pw[(r + 8 * g) * P_STRIDE + ln]      = (_Float16)p0;
            Pw[(r + 8 * g) * P_STRIDE + 16 + ln] = (_Float16)p1;
        }

        // ---- O += P * V ; l += P * ones  (A = probs from LDS, K=32 keys) ----
        union { v16h v; v8h h[2]; } ap;
        const _Float16* prow = Pw + ln * P_STRIDE;       // lane's query row M = ln
        ap.h[0] = *(const v8h*)(prow + 16 * g);          // keys 16g..16g+7
        ap.h[1] = *(const v8h*)(prow + 8 + 16 * g);      // keys 16g+8..16g+15
        {
            v16h bv0 = *(const v16h*)(Vt + (0 * 16 + ln) * VT_STRIDE + 16 * g);
            o0 = __builtin_amdgcn_wmma_f32_16x16x32_f16(false, ap.v, false, bv0, (short)0, o0, false, false);
            v16h bv1 = *(const v16h*)(Vt + (1 * 16 + ln) * VT_STRIDE + 16 * g);
            o1 = __builtin_amdgcn_wmma_f32_16x16x32_f16(false, ap.v, false, bv1, (short)0, o1, false, false);
            v16h bv2 = *(const v16h*)(Vt + (2 * 16 + ln) * VT_STRIDE + 16 * g);
            o2 = __builtin_amdgcn_wmma_f32_16x16x32_f16(false, ap.v, false, bv2, (short)0, o2, false, false);
            v16h bv3 = *(const v16h*)(Vt + (3 * 16 + ln) * VT_STRIDE + 16 * g);
            o3 = __builtin_amdgcn_wmma_f32_16x16x32_f16(false, ap.v, false, bv3, (short)0, o3, false, false);
            o4 = __builtin_amdgcn_wmma_f32_16x16x32_f16(false, ap.v, false, vones, (short)0, o4, false, false);
        }
    }

    // ---- normalize by l (= o4 row-sum) and store ----
    float* oh = out + headBase;
#pragma unroll
    for (int r = 0; r < 8; ++r) {
        int qpos = w * BLK + wv * 16 + r + 8 * g;
        if (qpos < T) {
            float inv = 1.0f / o4[r];
            float* orow = oh + (size_t)qpos * D + ln;
            orow[0]  = o0[r] * inv;
            orow[16] = o1[r] * inv;
            orow[32] = o2[r] * inv;
            orow[48] = o3[r] * inv;
        }
    }
}

// ---------------------------------------------------------------------------
// Row-0 kernel: query 0 attends to all T keys (overwrites out[:, :, 0, :]).
// One workgroup per (n,h); tiny (~33 MFLOP total), plain VALU.
// ---------------------------------------------------------------------------
__global__ __launch_bounds__(256) void bla_global_kernel(
    const float* __restrict__ q, const float* __restrict__ k,
    const float* __restrict__ v, const float* __restrict__ am,
    float* __restrict__ out)
{
    __shared__ float q0s[D];
    __shared__ float sbuf[T];
    __shared__ float red[8];
    __shared__ float ctxp[4][D];

    const int hh = blockIdx.x & (NH - 1);
    const int nn = blockIdx.x >> 4;
    const int tid = threadIdx.x;
    const size_t hb = ((size_t)(nn * NH + hh)) * T * D;
    const float* kh  = k + hb;
    const float* vh  = v + hb;
    const float* amn = am + (size_t)nn * T;

    if (tid < D) q0s[tid] = q[hb + tid];
    __syncthreads();

    // pass 1: scores + block max
    float lmax = FMIN;
    for (int key = tid; key < T; key += 256) {
        const float* kr = kh + (size_t)key * D;
        float acc = 0.f;
#pragma unroll
        for (int dd = 0; dd < D; ++dd) acc += q0s[dd] * kr[dd];
        float sc = acc * SCALE + amn[key];
        sbuf[key] = sc;
        lmax = fmaxf(lmax, sc);
    }
#pragma unroll
    for (int m = 1; m < 32; m <<= 1) lmax = fmaxf(lmax, __shfl_xor(lmax, m));
    if ((tid & 31) == 0) red[tid >> 5] = lmax;
    __syncthreads();
    float gmax = red[0];
#pragma unroll
    for (int i = 1; i < 8; ++i) gmax = fmaxf(gmax, red[i]);

    // pass 2: exp + block sum
    float lsum = 0.f;
    for (int key = tid; key < T; key += 256) {
        float p = __expf(sbuf[key] - gmax);
        sbuf[key] = p;
        lsum += p;
    }
#pragma unroll
    for (int m = 1; m < 32; m <<= 1) lsum += __shfl_xor(lsum, m);
    __syncthreads();                      // gmax reads done; sbuf p visible next
    if ((tid & 31) == 0) red[tid >> 5] = lsum;
    __syncthreads();
    float gsum = 0.f;
#pragma unroll
    for (int i = 0; i < 8; ++i) gsum += red[i];

    // pass 3: ctx[d] = sum_k p_k * V[k][d]; 4 partial threads per column
    const int col  = tid & 63;
    const int part = tid >> 6;            // 0..3
    float acc = 0.f;
    for (int key = part; key < T; key += 4)
        acc += sbuf[key] * vh[(size_t)key * D + col];
    ctxp[part][col] = acc;
    __syncthreads();
    if (part == 0) {
        float r2 = ctxp[0][col] + ctxp[1][col] + ctxp[2][col] + ctxp[3][col];
        out[hb + col] = r2 / gsum;
    }
}

extern "C" void kernel_launch(void* const* d_in, const int* in_sizes, int n_in,
                              void* d_out, int out_size, void* d_ws, size_t ws_size,
                              hipStream_t stream)
{
    (void)in_sizes; (void)n_in; (void)out_size; (void)d_ws; (void)ws_size;
    const float* q  = (const float*)d_in[0];
    const float* k  = (const float*)d_in[1];
    const float* v  = (const float*)d_in[2];
    const float* am = (const float*)d_in[3];
    float* out = (float*)d_out;

    dim3 grid(NW, NH, NB);
    bla_main_kernel<<<grid, 256, 0, stream>>>(q, k, v, am, out);
    bla_global_kernel<<<NB * NH, 256, 0, stream>>>(q, k, v, am, out);
}
